// MultiHeadAttention_33457795235982
// MI455X (gfx1250) — compile-verified, matrix operand order checked by reading, not run on hardware
//
#include <hip/hip_runtime.h>

// MI455X (gfx1250) multi-head causal attention, f16 WMMA with f32 accumulate.
// Pipeline: cvt_x -> cvt_w (transpose) -> gemm(Q) -> gemm(K) -> gemm(V^T)
//           -> flash attention (wave-per-16-query-rows) -> gemm(out, f32).
// Staging uses CDNA5 GLOBAL_LOAD_ASYNC_TO_LDS_B128 (ASYNCcnt) with software
// double buffering; fragments are assembled from LDS with ds_load_b128.
// Workspace layout (halves), total 48 MiB:
//   xh [4096,1024] | Wt 4x[1024,1024] | Qh [B,H,S,Dh] | Kh [B,H,S,Dh]
//   Vt [B,H,Dh,S]  | Oh [4096,1024]

#define D_MODEL 1024
#define SEQ     2048
#define BATCH   2
#define NH      16
#define DH      64
#define MTOT    (BATCH*SEQ)   // 4096

typedef __attribute__((ext_vector_type(16))) _Float16     v16h;
typedef __attribute__((ext_vector_type(4)))  _Float16     v4h;
typedef __attribute__((ext_vector_type(8)))  float        v8f;
typedef __attribute__((ext_vector_type(4)))  float        f32x4;
typedef __attribute__((ext_vector_type(4)))  unsigned int u32x4;

union Frag { v16h v; u32x4 u[2]; };

static __device__ __forceinline__ v8f wmma_f16(v16h a, v16h b, v8f c) {
  // D = A(16x32 f16) x B(32x16 f16) + C(16x16 f32)
  return __builtin_amdgcn_wmma_f32_16x16x32_f16(false, a, false, b, (short)0, c,
                                                false, false);
}

// Async DMA: 16B global -> LDS, tracked by ASYNCcnt (CDNA5 §10, opcode 98).
// LDS operand = wave-relative byte offset = low 32 bits of the generic
// address of a __shared__ object (shared aperture lives in addr[63:32]).
static __device__ __forceinline__ void async_b128(const _Float16* g,
                                                  _Float16* l) {
  unsigned lds = (unsigned)(uintptr_t)l;
  unsigned long long ga = (unsigned long long)(uintptr_t)g;
  asm volatile("global_load_async_to_lds_b128 %0, %1, off"
               :: "v"(lds), "v"(ga) : "memory");
}
template <int N>
static __device__ __forceinline__ void wait_async() {
  asm volatile("s_wait_asynccnt %0" :: "n"(N) : "memory");
}
static __device__ __forceinline__ void wait_ds0() {
  asm volatile("s_wait_dscnt 0" ::: "memory");
}

// ---------------------------------------------------------------- conversions
__global__ void cvt_x_kernel(const float* __restrict__ x,
                             _Float16* __restrict__ xh) {
  int i = blockIdx.x * blockDim.x + threadIdx.x;   // one float4 per thread
  f32x4 f = ((const f32x4*)x)[i];
  v4h h;
  h.x = (_Float16)f.x; h.y = (_Float16)f.y;
  h.z = (_Float16)f.z; h.w = (_Float16)f.w;
  ((v4h*)xh)[i] = h;
}

// Wt[w][e][d] = W[w][d][e], converted to f16 (output-column-major weights).
__global__ void cvt_w_kernel(const float* __restrict__ Wq,
                             const float* __restrict__ Wk,
                             const float* __restrict__ Wv,
                             const float* __restrict__ Wo,
                             _Float16* __restrict__ Wt) {
  __shared__ float tile[32][33];
  const float* W = blockIdx.z == 0 ? Wq : blockIdx.z == 1 ? Wk
                 : blockIdx.z == 2 ? Wv : Wo;
  _Float16* out = Wt + (size_t)blockIdx.z * D_MODEL * D_MODEL;
  int e0 = blockIdx.x * 32, d0 = blockIdx.y * 32;
  int tx = threadIdx.x;
  for (int r = threadIdx.y; r < 32; r += 8)
    tile[r][tx] = W[(size_t)(d0 + r) * D_MODEL + e0 + tx];
  __syncthreads();
  for (int r = threadIdx.y; r < 32; r += 8)
    out[(size_t)(e0 + r) * D_MODEL + d0 + tx] = (_Float16)tile[tx][r];
}

// ------------------------------------------------------------------ f16 GEMM
// Block: 256 threads = 8 waves; tile 128(M) x 64(N); K-steps of 32.
// Double-buffered LDS (pitch 40 halves, conflict-free b128), filled by
// async-to-LDS DMA: issue(next tile, 3 ops/thread) -> s_wait_asynccnt 3
// -> barrier -> 4 WMMA/wave -> s_wait_dscnt 0 -> barrier.
// mode 0/1: Q/K scatter to [B,H,S,Dh] f16;  mode 2: V scatter to [B,H,Dh,S];
// mode 3: f32 output [M, D_MODEL].
__global__ __launch_bounds__(256) void gemm_kernel(
    const _Float16* __restrict__ A, const _Float16* __restrict__ Wt,
    const float* __restrict__ bias, void* __restrict__ out, int mode) {
  __shared__ __align__(16) _Float16 As[2][128 * 40];
  __shared__ __align__(16) _Float16 Bs[2][64 * 40];
  const int tid = threadIdx.x;
  const int lane = tid & 31, wave = tid >> 5;
  const int hf = lane >> 4, ln = lane & 15;
  const int m0 = blockIdx.y * 128, n0 = blockIdx.x * 64;

  v8f acc[4] = {};

  auto stage = [&](int k0, int buf) {
#pragma unroll
    for (int rep = 0; rep < 2; ++rep) {           // 512 x 16B of A
      int id = tid + rep * 256;
      int row = id >> 2, c = id & 3;
      async_b128(A + (size_t)(m0 + row) * D_MODEL + k0 + c * 8,
                 As[buf] + row * 40 + c * 8);
    }
    int row = tid >> 2, c = tid & 3;              // 256 x 16B of B
    async_b128(Wt + (size_t)(n0 + row) * D_MODEL + k0 + c * 8,
               Bs[buf] + row * 40 + c * 8);
  };

  stage(0, 0);
  for (int k0 = 0, it = 0; k0 < D_MODEL; k0 += 32, ++it) {
    const int cur = it & 1;
    const bool more = (k0 + 32 < D_MODEL);
    if (more) {
      stage(k0 + 32, cur ^ 1);
      __builtin_prefetch(A + (size_t)(m0 + (tid >> 1)) * D_MODEL + k0 + 64 +
                             (tid & 1) * 16, 0, 1);
      wait_async<3>();                            // current tile landed
    } else {
      wait_async<0>();
    }
    __syncthreads();

    Frag af;                                      // A: 16x32, rows = lane&15
    const u32x4* ap = (const u32x4*)(As[cur] + (wave * 16 + ln) * 40);
    af.u[0] = ap[hf]; af.u[1] = ap[2 + hf];
#pragma unroll
    for (int nt = 0; nt < 4; ++nt) {
      Frag bf;
      const u32x4* bp = (const u32x4*)(Bs[cur] + (nt * 16 + ln) * 40);
      bf.u[0] = bp[hf]; bf.u[1] = bp[2 + hf];
      acc[nt] = wmma_f16(af.v, bf.v, acc[nt]);
    }
    wait_ds0();                                   // LDS reads done before
    __syncthreads();                              // next DMA overwrites
  }

  // bias + store. C layout: VGPR r -> row r (lanes 0-15) / r+8 (lanes 16-31).
  if (mode < 3) {
    _Float16* dst = (_Float16*)out;
#pragma unroll
    for (int nt = 0; nt < 4; ++nt) {
      int e = n0 + nt * 16 + ln;
      float bv = bias[e];
      int h = e >> 6, dh = e & 63;
#pragma unroll
      for (int r = 0; r < 8; ++r) {
        int mg = m0 + wave * 16 + r + hf * 8;
        int b = mg >> 11, s = mg & 2047;
        float val = acc[nt][r] + bv;
        size_t idx = (mode == 2)
            ? ((size_t)((b * NH + h) * DH + dh)) * SEQ + s       // V^T
            : ((size_t)((b * NH + h) * SEQ + s)) * DH + dh;      // Q/K
        dst[idx] = (_Float16)val;
      }
    }
  } else {
    float* dst = (float*)out;
#pragma unroll
    for (int nt = 0; nt < 4; ++nt) {
      int e = n0 + nt * 16 + ln;
      float bv = bias[e];
#pragma unroll
      for (int r = 0; r < 8; ++r) {
        int mg = m0 + wave * 16 + r + hf * 8;
        dst[(size_t)mg * D_MODEL + e] = acc[nt][r] + bv;
      }
    }
  }
}

// ------------------------------------------------------------ flash attention
// One wave per 16 query rows per (b,h). Chunks of 32 keys, causal, online
// softmax in exp2 domain. K/V staged by async-to-LDS DMA, double buffered:
// 16 async ops per chunk, s_wait_asynccnt 16 overlaps next chunk's DMA with
// the current chunk's WMMAs. Single-wave workgroup; explicit s_wait_dscnt
// orders fragment ds_loads against DMA LDS writes (ASYNC/DS pipes are
// mutually unordered).
__global__ __launch_bounds__(32) void attn_kernel(
    const _Float16* __restrict__ Qh, const _Float16* __restrict__ Kh,
    const _Float16* __restrict__ Vt, _Float16* __restrict__ Oh) {
  __shared__ __align__(16) _Float16 qb[16 * 72];      // 16x64, pitch 72
  __shared__ __align__(16) _Float16 kb[2][32 * 72];   // 32x64, pitch 72
  __shared__ __align__(16) _Float16 vb[2][64 * 40];   // 64(dh)x32(t), pitch 40
  __shared__ __align__(16) _Float16 pb[16 * 40];      // 16x32 probs, pitch 40

  const int lane = threadIdx.x;
  const int hf = lane >> 4, ln = lane & 15;
  const int qs0 = blockIdx.x * 16;
  const int bh = blockIdx.y;                       // 0..31
  const int b = bh >> 4, h = bh & 15;

  const _Float16* Qbase = Qh + ((size_t)bh * SEQ + qs0) * DH;
  const _Float16* Kbase = Kh + (size_t)bh * SEQ * DH;
  const _Float16* Vbase = Vt + (size_t)bh * DH * SEQ;

#pragma unroll
  for (int t = 0; t < 4; ++t) {                    // stage Q tile (4 DMA ops)
    int id = lane + t * 32;
    int row = id >> 3, c = id & 7;
    async_b128(Qbase + (size_t)row * DH + c * 8, qb + row * 72 + c * 8);
  }
  wait_async<0>();

  Frag qf[2];
#pragma unroll
  for (int kk = 0; kk < 2; ++kk) {                 // Dh split into two K=32
    const u32x4* qp = (const u32x4*)(qb + ln * 72 + kk * 32);
    qf[kk].u[0] = qp[hf]; qf[kk].u[1] = qp[2 + hf];
  }

  auto stage_kv = [&](int t0, int buf) {
#pragma unroll
    for (int t = 0; t < 8; ++t) {                  // K chunk 32x64 (8 ops)
      int id = lane + t * 32;
      int row = id >> 3, c = id & 7;
      async_b128(Kbase + (size_t)(t0 + row) * DH + c * 8,
                 kb[buf] + row * 72 + c * 8);
    }
#pragma unroll
    for (int t = 0; t < 8; ++t) {                  // V^T chunk 64x32 (8 ops)
      int id = lane + t * 32;
      int row = id >> 2, c = id & 3;
      async_b128(Vbase + (size_t)row * SEQ + t0 + c * 8,
                 vb[buf] + row * 40 + c * 8);
    }
  };

  float m_r[8], l_r[8];
  v8f o[4] = {};
  const float NEG = -__builtin_inff();
#pragma unroll
  for (int r = 0; r < 8; ++r) { m_r[r] = NEG; l_r[r] = 0.f; }

  const float SC = 0.125f * 1.44269504088896340736f;  // Dh^-0.5 * log2(e)
  const int jmax = (qs0 + 15) >> 5;

  stage_kv(0, 0);
  for (int j = 0; j <= jmax; ++j) {
    const int t0 = j * 32;
    const int cur = j & 1;
    if (j < jmax) {
      wait_ds0();                                  // prior reads of buf cur^1
      stage_kv(t0 + 32, cur ^ 1);
      wait_async<16>();                            // current chunk landed
    } else {
      wait_async<0>();
    }

    v8f dots[2] = {};
#pragma unroll
    for (int nt = 0; nt < 2; ++nt) {               // S = Q K^T
#pragma unroll
      for (int kk = 0; kk < 2; ++kk) {
        Frag bf;
        const u32x4* bp =
            (const u32x4*)(kb[cur] + (nt * 16 + ln) * 72 + kk * 32);
        bf.u[0] = bp[hf]; bf.u[1] = bp[2 + hf];
        dots[nt] = wmma_f16(qf[kk].v, bf.v, dots[nt]);
      }
    }

#pragma unroll
    for (int r = 0; r < 8; ++r) {                  // online softmax per row
      int qrow = qs0 + r + hf * 8;
      float s0 = dots[0][r] * SC;
      float s1 = dots[1][r] * SC;
      if (t0 + ln      > qrow) s0 = NEG;
      if (t0 + 16 + ln > qrow) s1 = NEG;
      float mx = fmaxf(s0, s1);
      mx = fmaxf(mx, __shfl_xor(mx, 1, 16));
      mx = fmaxf(mx, __shfl_xor(mx, 2, 16));
      mx = fmaxf(mx, __shfl_xor(mx, 4, 16));
      mx = fmaxf(mx, __shfl_xor(mx, 8, 16));
      float mn  = fmaxf(m_r[r], mx);
      float cor = exp2f(m_r[r] - mn);
      float p0  = exp2f(s0 - mn);
      float p1  = exp2f(s1 - mn);
      float rs = p0 + p1;
      rs += __shfl_xor(rs, 1, 16);
      rs += __shfl_xor(rs, 2, 16);
      rs += __shfl_xor(rs, 4, 16);
      rs += __shfl_xor(rs, 8, 16);
      l_r[r] = l_r[r] * cor + rs;
      m_r[r] = mn;
#pragma unroll
      for (int nt = 0; nt < 4; ++nt) o[nt][r] *= cor;
      int prow = r + hf * 8;
      pb[prow * 40 + ln]      = (_Float16)p0;      // C-layout -> LDS
      pb[prow * 40 + 16 + ln] = (_Float16)p1;
    }
    wait_ds0();

    Frag pf;                                       // LDS -> A-fragment layout
    const u32x4* pp = (const u32x4*)(pb + ln * 40);
    pf.u[0] = pp[hf]; pf.u[1] = pp[2 + hf];
#pragma unroll
    for (int nt = 0; nt < 4; ++nt) {               // O += P V
      Frag vf;
      const u32x4* vp = (const u32x4*)(vb[cur] + (nt * 16 + ln) * 40);
      vf.u[0] = vp[hf]; vf.u[1] = vp[2 + hf];
      o[nt] = wmma_f16(pf.v, vf.v, o[nt]);
    }
  }

#pragma unroll
  for (int nt = 0; nt < 4; ++nt) {                 // normalize + store [B,S,D]
#pragma unroll
    for (int r = 0; r < 8; ++r) {
      int s = qs0 + r + hf * 8;
      int dh = nt * 16 + ln;
      float val = o[nt][r] / l_r[r];
      Oh[((size_t)(b * SEQ + s)) * D_MODEL + h * DH + dh] = (_Float16)val;
    }
  }
}

// -------------------------------------------------------------------- launch
extern "C" void kernel_launch(void* const* d_in, const int* in_sizes, int n_in,
                              void* d_out, int out_size, void* d_ws,
                              size_t ws_size, hipStream_t stream) {
  (void)in_sizes; (void)n_in; (void)out_size; (void)ws_size;
  const float* x  = (const float*)d_in[0];
  const float* Wq = (const float*)d_in[1];
  const float* bq = (const float*)d_in[2];
  const float* Wk = (const float*)d_in[3];
  const float* bk = (const float*)d_in[4];
  const float* Wv = (const float*)d_in[5];
  const float* bv = (const float*)d_in[6];
  const float* Wo = (const float*)d_in[7];
  const float* bo = (const float*)d_in[8];

  _Float16* xh = (_Float16*)d_ws;
  _Float16* Wt = xh + (size_t)MTOT * D_MODEL;
  _Float16* Qh = Wt + (size_t)4 * D_MODEL * D_MODEL;
  _Float16* Kh = Qh + (size_t)BATCH * NH * SEQ * DH;
  _Float16* Vt = Kh + (size_t)BATCH * NH * SEQ * DH;
  _Float16* Oh = Vt + (size_t)BATCH * NH * SEQ * DH;

  cvt_x_kernel<<<dim3((MTOT * D_MODEL / 4) / 256), 256, 0, stream>>>(x, xh);
  cvt_w_kernel<<<dim3(32, 32, 4), dim3(32, 8), 0, stream>>>(Wq, Wk, Wv, Wo, Wt);

  dim3 gg(D_MODEL / 64, MTOT / 128);
  gemm_kernel<<<gg, 256, 0, stream>>>(xh, Wt + (size_t)0 * D_MODEL * D_MODEL,
                                      bq, (void*)Qh, 0);
  gemm_kernel<<<gg, 256, 0, stream>>>(xh, Wt + (size_t)1 * D_MODEL * D_MODEL,
                                      bk, (void*)Kh, 1);
  gemm_kernel<<<gg, 256, 0, stream>>>(xh, Wt + (size_t)2 * D_MODEL * D_MODEL,
                                      bv, (void*)Vt, 2);
  attn_kernel<<<dim3(SEQ / 16, BATCH * NH), 32, 0, stream>>>(Qh, Kh, Vt, Oh);
  gemm_kernel<<<gg, 256, 0, stream>>>(Oh, Wt + (size_t)3 * D_MODEL * D_MODEL,
                                      bo, d_out, 3);
}
